// GroupedQueryAttention_8194797601119
// MI455X (gfx1250) — matrix operand order repaired and checked by reading, not compiled, not run on hardware
//
#include <hip/hip_runtime.h>

// ---------------------------------------------------------------------------
// Problem constants (match reference)
// ---------------------------------------------------------------------------
static constexpr int Bc  = 2;
static constexpr int Sc  = 2048;
static constexpr int Dm  = 2048;   // d_model
static constexpr int Hc  = 32;     // n_heads
static constexpr int Hkv = 8;      // n_kv_heads
static constexpr int Dh  = 64;     // head_dim

// ---------------------------------------------------------------------------
// CDNA5 WMMA plumbing (gfx1250, wave32)
// ---------------------------------------------------------------------------
typedef __bf16 bf16_t;
typedef __attribute__((ext_vector_type(16))) __bf16        v16bf;
typedef __attribute__((ext_vector_type(8)))  float         v8f;
typedef __attribute__((ext_vector_type(4)))  unsigned int  u32x4;

union Frag16 { u32x4 u[2]; v16bf v; };

// A-matrix 16x32 bf16 fragment (ISA 7.12.2): lane holds row M = lane&15.
// lanes 0-15: K = {0..7,16..23}; lanes 16-31: K = {8..15,24..31}.
__device__ inline v16bf load_a_frag(const bf16_t* __restrict__ row, int lane) {
  const int kb = (lane & 16) ? 8 : 0;
  Frag16 f;
  f.u[0] = *(const u32x4*)(row + kb);        // K = kb .. kb+7
  f.u[1] = *(const u32x4*)(row + kb + 16);   // K = kb+16 .. kb+23
  return f.v;
}

// B-matrix 32x16 bf16 fragment: lane holds column N = lane&15.
// lanes 0-15: K = {0..7,16..23}; lanes 16-31: K = {8..15,24..31}.
__device__ inline v16bf load_b_frag(const bf16_t* __restrict__ row, int lane) {
  const int kb = (lane & 16) ? 8 : 0;
  Frag16 f;
  f.u[0] = *(const u32x4*)(row + kb);
  f.u[1] = *(const u32x4*)(row + kb + 16);
  return f.v;
}

__device__ inline v8f wmma_bf16(v16bf a, v16bf b, v8f c) {
  return __builtin_amdgcn_wmma_f32_16x16x32_bf16(
      /*neg_a=*/false, a, /*neg_b=*/false, b,
      /*c_mod=*/(short)0, c, /*reuse_a=*/false, /*reuse_b=*/false);
}

// ---------------------------------------------------------------------------
// CDNA5 async copy: global -> LDS (ASYNCcnt-tracked). INST_OFFSET applies to
// both the LDS and global addresses, so one base pair covers 32 bytes.
// LDS dest address = low 32 bits of the generic shared pointer.
// ---------------------------------------------------------------------------
__device__ inline void async_b128x2(unsigned lds_addr, const bf16_t* gptr) {
  asm volatile("global_load_async_to_lds_b128 %0, %1, off\n\t"
               "global_load_async_to_lds_b128 %0, %1, off offset:16"
               :: "v"(lds_addr), "v"(gptr) : "memory");
}
__device__ inline void wait_async_le4() {
  asm volatile("s_wait_asynccnt 0x4" ::: "memory");
}
__device__ inline void wait_async_0() {
  asm volatile("s_wait_asynccnt 0x0" ::: "memory");
}

// ---------------------------------------------------------------------------
// Elementwise helpers
// ---------------------------------------------------------------------------
__global__ void cast_f32_bf16(const float* __restrict__ src,
                              bf16_t* __restrict__ dst, int n) {
  int i = blockIdx.x * blockDim.x + threadIdx.x;
  if (i < n) dst[i] = (bf16_t)src[i];
}

// RoPE (rotate_half variant of the reference) fused with f32->bf16 cast.
// X layout: [B, S, nheads, 64]. cos/sin: [S, 32].
__global__ void rope_cast(const float* __restrict__ X,
                          const float* __restrict__ cosT,
                          const float* __restrict__ sinT,
                          bf16_t* __restrict__ Y,
                          int nheads, float scale, int total) {
  int i = blockIdx.x * blockDim.x + threadIdx.x;
  if (i >= total) return;
  int d = i & 63;
  int s = (i / (64 * nheads)) & (Sc - 1);
  float c  = cosT[s * 32 + (d >> 1)];
  float sn = sinT[s * 32 + (d >> 1)];
  float v  = X[i];
  float vr = (d < 32) ? -X[i + 32] : X[i - 32];
  Y[i] = (bf16_t)((v * c + vr * sn) * scale);
}

// V: [B,S,Hkv,64] f32  ->  V^T: [B,Hkv,64,S] bf16 (contiguous tokens per dim)
__global__ void transpose_v(const float* __restrict__ Vf,
                            bf16_t* __restrict__ Vt) {
  int i = blockIdx.x * blockDim.x + threadIdx.x;
  if (i >= Bc * Sc * Hkv * Dh) return;
  int d   = i & 63;
  int hkv = (i >> 6) & (Hkv - 1);
  int s   = (i >> 9) & (Sc - 1);
  int b   = i >> 20;                     // S*Hkv*Dh = 2^20 per batch
  Vt[(((size_t)b * Hkv + hkv) * Dh + d) * Sc + s] = (bf16_t)Vf[i];
}

// ---------------------------------------------------------------------------
// Tiled WMMA GEMM with double-buffered async-LDS staging:
//   C[M,N] = A[M,K] @ W[N,K]^T, bf16 in, f32 out.
// Block = 256 threads (8 waves), block tile 128x128, K-step 32.
// Wave (wm in 0..3, wn in 0..1) computes a 32x64 C tile: 8 WMMAs / K-step.
// A/B tiles (8 KB each) staged via global_load_async_to_lds_b128, ping-pong.
// Steady-state loop: stage(k+1); s_wait_asynccnt<=4; barrier; 8x WMMA; barrier.
// grid = (N/128, M/128)
// ---------------------------------------------------------------------------
__global__ __launch_bounds__(256)
void gemm_bf16_wmma(const bf16_t* __restrict__ A,
                    const bf16_t* __restrict__ W,
                    float* __restrict__ C, int M, int N, int K) {
  // [buf][ A(128x32) | B(128x32) ]
  __shared__ __align__(16) bf16_t sh[2][2 * 128 * 32];

  const int tid  = threadIdx.x;
  const int lane = tid & 31;
  const int wid  = tid >> 5;
  const int wm   = wid & 3;        // 0..3  -> 32-row slice
  const int wn   = wid >> 2;       // 0..1  -> 64-col slice
  const int cl   = lane & 15;
  const int rb   = (lane & 16) ? 8 : 0;

  const int n0b = blockIdx.x * 128;
  const int m0b = blockIdx.y * 128;

  // ---- async stage of one (A,B) K-step tile: 4 x b128 per thread ----
  const int srow = tid >> 1;              // 0..127
  const int ske  = (tid & 1) * 16;        // 0 or 16 elements into the K-step
  const bf16_t* gA = A + (size_t)(m0b + srow) * K + ske;
  const bf16_t* gB = W + (size_t)(n0b + srow) * K + ske;
  const unsigned ldsA0 = (unsigned)(size_t)(&sh[0][0])        + (unsigned)tid * 32u;
  const unsigned ldsB0 = (unsigned)(size_t)(&sh[0][128 * 32]) + (unsigned)tid * 32u;
  const unsigned bufStride = (unsigned)(2 * 128 * 32 * sizeof(bf16_t));

  auto stage = [&](int k0, int buf) {
    const unsigned boff = (unsigned)buf * bufStride;
    async_b128x2(ldsA0 + boff, gA + k0);
    async_b128x2(ldsB0 + boff, gB + k0);
  };

  v8f acc[2][4] = {};

  auto compute = [&](int buf) {
    const bf16_t* ab = &sh[buf][0];
    const bf16_t* bb = &sh[buf][128 * 32];
    const v16bf a0 = load_a_frag(ab + (wm * 32 + cl) * 32, lane);
    const v16bf a1 = load_a_frag(ab + (wm * 32 + 16 + cl) * 32, lane);
#pragma unroll
    for (int nt = 0; nt < 4; ++nt) {
      const v16bf b = load_b_frag(bb + (wn * 64 + nt * 16 + cl) * 32, lane);
      acc[0][nt] = wmma_bf16(a0, b, acc[0][nt]);
      acc[1][nt] = wmma_bf16(a1, b, acc[1][nt]);
    }
  };

  const int KT = K >> 5;
  stage(0, 0);
  // steady state: unconditional wait<=4 keeps fetch of tile k+1 in flight
  // while tile k is consumed.
  for (int kt = 0; kt < KT - 1; ++kt) {
    const int cur = kt & 1;
    stage((kt + 1) << 5, cur ^ 1);
    wait_async_le4();            // tile kt's 4 loads complete (in-order)
    __syncthreads();             // all threads' async writes visible
    compute(cur);
    __syncthreads();             // done reading before next overwrite
  }
  // last tile (no further staging)
  wait_async_0();
  __syncthreads();
  compute((KT - 1) & 1);

#pragma unroll
  for (int mt = 0; mt < 2; ++mt) {
#pragma unroll
    for (int nt = 0; nt < 4; ++nt) {
      float* cp = C + (size_t)(m0b + wm * 32 + mt * 16 + rb) * N
                    + n0b + wn * 64 + nt * 16 + cl;
#pragma unroll
      for (int r = 0; r < 8; ++r) cp[(size_t)r * N] = acc[mt][nt][r];
    }
  }
}

// ---------------------------------------------------------------------------
// Flash attention (causal, GQA): one wave per (b, h, 16-query tile).
// Q: [B,S,32,64] bf16 (pre-scaled by 1/8), K: [B,S,8,64] bf16,
// Vt: [B,8,64,S] bf16, O: [B,S,2048] bf16.
// 32 keys per iteration: 4 WMMA (QK^T) + 4 WMMA (PV).
// ---------------------------------------------------------------------------
__global__ __launch_bounds__(256)
void flash_attn(const bf16_t* __restrict__ Q, const bf16_t* __restrict__ K,
                const bf16_t* __restrict__ Vt, bf16_t* __restrict__ O) {
  __shared__ __align__(16) bf16_t lds_p[8][16 * 32];

  const int lane = threadIdx.x & 31;
  const int wid  = threadIdx.x >> 5;
  const int g    = blockIdx.x * 8 + wid;     // 0 .. B*Hc*(S/16)-1 = 8191
  const int qt   = g & 127;                  // S/16 = 128
  const int h    = (g >> 7) & 31;
  const int b    = g >> 12;
  const int hkv  = h >> 2;
  const int q0   = qt * 16;
  const int cl   = lane & 15;
  const int rb   = (lane & 16) ? 8 : 0;

  const bf16_t* Qrow  = Q + (((size_t)b * Sc + q0 + cl) * Hc + h) * Dh;
  const v16bf qa0 = load_a_frag(Qrow, lane);        // dims 0..31
  const v16bf qa1 = load_a_frag(Qrow + 32, lane);   // dims 32..63

  const bf16_t* Kbase = K + ((size_t)b * Sc * Hkv + hkv) * Dh;    // + t*Hkv*Dh
  const bf16_t* Vbase = Vt + ((size_t)(b * Hkv + hkv) * Dh) * Sc; // + d*Sc + t

  v8f o0 = {}, o1 = {}, o2 = {}, o3 = {};
  float m[8], l[8];
#pragma unroll
  for (int r = 0; r < 8; ++r) { m[r] = -3.0e38f; l[r] = 0.f; }

  const int kend = q0 + 16;
  for (int kc = 0; kc < kend; kc += 32) {
    // ---- scores S = Q @ K^T (16x32) ----
    v8f s0 = {}, s1 = {};
    {
      const bf16_t* kr0 = Kbase + (size_t)(kc + cl) * (Hkv * Dh);
      s0 = wmma_bf16(qa0, load_b_frag(kr0, lane), s0);
      s0 = wmma_bf16(qa1, load_b_frag(kr0 + 32, lane), s0);
      const bf16_t* kr1 = Kbase + (size_t)(kc + 16 + cl) * (Hkv * Dh);
      s1 = wmma_bf16(qa0, load_b_frag(kr1, lane), s1);
      s1 = wmma_bf16(qa1, load_b_frag(kr1 + 32, lane), s1);
    }
    // ---- causal mask (only the straddling chunks need it) ----
    if (kc + 31 > q0) {
#pragma unroll
      for (int r = 0; r < 8; ++r) {
        const int q = q0 + rb + r;
        if (kc + cl > q)      s0[r] = -3.0e38f;
        if (kc + 16 + cl > q) s1[r] = -3.0e38f;
      }
    }
    // ---- online softmax; P -> LDS as bf16 in A-frag friendly layout ----
    float alpha[8];
#pragma unroll
    for (int r = 0; r < 8; ++r) {
      float mx = fmaxf(s0[r], s1[r]);
      mx = fmaxf(mx, __shfl_xor(mx, 1));
      mx = fmaxf(mx, __shfl_xor(mx, 2));
      mx = fmaxf(mx, __shfl_xor(mx, 4));
      mx = fmaxf(mx, __shfl_xor(mx, 8));
      const float mn = fmaxf(m[r], mx);
      const float a  = __expf(m[r] - mn);
      const float p0 = __expf(s0[r] - mn);
      const float p1 = __expf(s1[r] - mn);
      float rs = p0 + p1;
      rs += __shfl_xor(rs, 1);
      rs += __shfl_xor(rs, 2);
      rs += __shfl_xor(rs, 4);
      rs += __shfl_xor(rs, 8);
      l[r] = l[r] * a + rs;
      m[r] = mn;
      alpha[r] = a;
      const int row = rb + r;
      lds_p[wid][row * 32 + cl]      = (bf16_t)p0;
      lds_p[wid][row * 32 + 16 + cl] = (bf16_t)p1;
    }
    // ---- rescale running O, then O += P @ V ----
#pragma unroll
    for (int r = 0; r < 8; ++r) {
      o0[r] *= alpha[r]; o1[r] *= alpha[r];
      o2[r] *= alpha[r]; o3[r] *= alpha[r];
    }
    // same-wave LDS ops are in-order: safe to reload without a barrier
    const v16bf pa = load_a_frag(&lds_p[wid][cl * 32], lane);
    o0 = wmma_bf16(pa, load_b_frag(Vbase + (size_t)(cl)      * Sc + kc, lane), o0);
    o1 = wmma_bf16(pa, load_b_frag(Vbase + (size_t)(16 + cl) * Sc + kc, lane), o1);
    o2 = wmma_bf16(pa, load_b_frag(Vbase + (size_t)(32 + cl) * Sc + kc, lane), o2);
    o3 = wmma_bf16(pa, load_b_frag(Vbase + (size_t)(48 + cl) * Sc + kc, lane), o3);
  }

  // ---- finalize: divide by row sums, store bf16 [B,S,H*D] ----
#pragma unroll
  for (int r = 0; r < 8; ++r) {
    const float inv = 1.0f / l[r];
    bf16_t* p = O + ((size_t)b * Sc + q0 + rb + r) * (size_t)Dm + h * Dh + cl;
    p[0]  = (bf16_t)(o0[r] * inv);
    p[16] = (bf16_t)(o1[r] * inv);
    p[32] = (bf16_t)(o2[r] * inv);
    p[48] = (bf16_t)(o3[r] * inv);
  }
}

// ---------------------------------------------------------------------------
// Launch
// ---------------------------------------------------------------------------
extern "C" void kernel_launch(void* const* d_in, const int* in_sizes, int n_in,
                              void* d_out, int out_size, void* d_ws, size_t ws_size,
                              hipStream_t stream) {
  const float* x    = (const float*)d_in[0];
  const float* wq   = (const float*)d_in[1];
  const float* wk   = (const float*)d_in[2];
  const float* wv   = (const float*)d_in[3];
  const float* wo   = (const float*)d_in[4];
  const float* cosT = (const float*)d_in[5];
  const float* sinT = (const float*)d_in[6];
  float* out = (float*)d_out;

  const size_t nX  = (size_t)Bc * Sc * Dm;          // 8,388,608
  const size_t nWq = (size_t)Hc * Dh * Dm;          // 4,194,304
  const size_t nWk = (size_t)Hkv * Dh * Dm;         // 1,048,576
  const size_t nQ  = (size_t)Bc * Sc * Hc * Dh;     // 8,388,608
  const size_t nK  = (size_t)Bc * Sc * Hkv * Dh;    // 2,097,152

  char* ws = (char*)d_ws;
  size_t off = 0;
  auto alloc = [&](size_t bytes) -> void* {
    void* p = ws + off;
    off = (off + bytes + 255) & ~(size_t)255;
    return p;
  };
  bf16_t* xb  = (bf16_t*)alloc(nX * 2);
  bf16_t* wqb = (bf16_t*)alloc(nWq * 2);
  bf16_t* wkb = (bf16_t*)alloc(nWk * 2);
  bf16_t* wvb = (bf16_t*)alloc(nWk * 2);
  bf16_t* wob = (bf16_t*)alloc(nWq * 2);
  float*  Qf  = (float*)alloc(nQ * 4);
  float*  Kf  = (float*)alloc(nK * 4);
  float*  Vf  = (float*)alloc(nK * 4);
  bf16_t* Qb  = (bf16_t*)alloc(nQ * 2);
  bf16_t* Kb  = (bf16_t*)alloc(nK * 2);
  bf16_t* Vt  = (bf16_t*)alloc(nK * 2);
  bf16_t* Ob  = (bf16_t*)alloc(nQ * 2);
  (void)in_sizes; (void)n_in; (void)out_size; (void)ws_size;

  const int TB = 256;
  // 1) casts to bf16
  cast_f32_bf16<<<(int)((nX  + TB - 1) / TB), TB, 0, stream>>>(x,  xb,  (int)nX);
  cast_f32_bf16<<<(int)((nWq + TB - 1) / TB), TB, 0, stream>>>(wq, wqb, (int)nWq);
  cast_f32_bf16<<<(int)((nWk + TB - 1) / TB), TB, 0, stream>>>(wk, wkb, (int)nWk);
  cast_f32_bf16<<<(int)((nWk + TB - 1) / TB), TB, 0, stream>>>(wv, wvb, (int)nWk);
  cast_f32_bf16<<<(int)((nWq + TB - 1) / TB), TB, 0, stream>>>(wo, wob, (int)nWq);

  // 2) QKV projections (WMMA + async-LDS pipeline)
  const int Mrows = Bc * Sc; // 4096
  gemm_bf16_wmma<<<dim3(Dm / 128,         Mrows / 128), 256, 0, stream>>>(xb, wqb, Qf, Mrows, Dm,       Dm);
  gemm_bf16_wmma<<<dim3((Hkv * Dh) / 128, Mrows / 128), 256, 0, stream>>>(xb, wkb, Kf, Mrows, Hkv * Dh, Dm);
  gemm_bf16_wmma<<<dim3((Hkv * Dh) / 128, Mrows / 128), 256, 0, stream>>>(xb, wvb, Vf, Mrows, Hkv * Dh, Dm);

  // 3) RoPE + casts (+ fold in 1/sqrt(head_dim) on Q), V transpose
  rope_cast<<<(int)(nQ / TB), TB, 0, stream>>>(Qf, cosT, sinT, Qb, Hc,  0.125f, (int)nQ);
  rope_cast<<<(int)(nK / TB), TB, 0, stream>>>(Kf, cosT, sinT, Kb, Hkv, 1.0f,   (int)nK);
  transpose_v<<<(int)(nK / TB), TB, 0, stream>>>(Vf, Vt);

  // 4) causal GQA flash attention (WMMA)
  flash_attn<<<(Bc * Hc * (Sc / 16)) / 8, 256, 0, stream>>>(Qb, Kb, Vt, Ob);

  // 5) output projection (WMMA) -> f32 d_out
  gemm_bf16_wmma<<<dim3(Dm / 128, Mrows / 128), 256, 0, stream>>>(Ob, wob, out, Mrows, Dm, Dm);
}
